// Cross_Attention_Global_5952824672939
// MI455X (gfx1250) — compile-verified, matrix operand order checked by reading, not run on hardware
//
#include <hip/hip_runtime.h>

// ---------------------------------------------------------------------------
// Cross attention (bidirectional, global scores + InstanceNorm + softmax)
// B=8 per direction, N=2048, C=64, CH=512, J = B*CH = 4096.
// - All big GEMMs on v_wmma_f32_16x16x32_f16 (f16 in, f32 accumulate)
// - Fragments = two aligned 16-byte LDS reads per lane (ds_load_b128)
// - f16 operand tiles staged with GLOBAL_LOAD_ASYNC_TO_LDS_B128 (ASYNCcnt)
// - Double-buffered LDS pipeline: tile i+1 staged (async DMA / cvt stores)
//   while tile i runs on the WMMA pipes.
// - Q16/K16 stored transposed ([sample][ch][n]) so the dominant scores GEMM
//   stages both tiles with direct async copies (no LDS-side transpose).
// ---------------------------------------------------------------------------

typedef __attribute__((ext_vector_type(16))) _Float16 v16h;
typedef __attribute__((ext_vector_type(8)))  _Float16 v8h;
typedef __attribute__((ext_vector_type(8)))  float    v8f;

#define C_DIM   64
#define CH_DIM  512
#define N_SEQ   2048
#define NJ      4096          // 8 * CH_DIM
#define EPS_IN  1e-5f

#define LDK 40                // LDS row stride in halfs (80 B, 16B-aligned rows)

// Per-lane WMMA 16-bit fragment: two contiguous 16-byte chunks of the row.
// A (16x32): lane l -> row m=l&15, h=l>>4; halfs [h*8 .. h*8+7] and [16+h*8 ..].
// B staged transposed ([col][k]) uses the identical gather.
__device__ inline v16h frag_lds(const _Float16* rowbase, int h) {
  const v8h lo = *(const v8h*)(rowbase + (h << 3));
  const v8h hi = *(const v8h*)(rowbase + 16 + (h << 3));
  return __builtin_shufflevector(lo, hi, 0, 1, 2, 3, 4, 5, 6, 7,
                                 8, 9, 10, 11, 12, 13, 14, 15);
}

// Async global->LDS copy of 16 bytes per lane (CDNA5, tracked by ASYNCcnt).
__device__ inline void async_g2l_b128(_Float16* lds, const _Float16* g) {
  asm volatile("global_load_async_to_lds_b128 %0, %1, off"
               :: "v"((unsigned)(unsigned long long)lds), "v"(g)
               : "memory");
}
__device__ inline void async_wait_all() {
  asm volatile("s_wait_asynccnt 0x0" ::: "memory");
}

// ---------------------------------------------------------------------------
// K1: QKV projections.  Q/K written TRANSPOSED: [sample][ch][n] (n contiguous)
//     V written row-major: [sample][n][ch].
// M = 16*2048 flat, K = 64, N = 512.  Block tile 128x64.  grid = (8, 256, 3)
// ---------------------------------------------------------------------------
__global__ __launch_bounds__(128)
void proj_gemm_kernel(const float* __restrict__ emb, const float* __restrict__ Wq,
                      const float* __restrict__ Wk, const float* __restrict__ Wv,
                      _Float16* __restrict__ Q16, _Float16* __restrict__ K16,
                      _Float16* __restrict__ V16)
{
  __shared__ __align__(16) _Float16 lA[2][128 * LDK];   // [m][k]
  __shared__ __align__(16) _Float16 lBt[2][64 * LDK];   // [n][k] (B transposed)
  const int tid = threadIdx.x, lane = tid & 31, wave = tid >> 5, h = lane >> 4;
  const int ch0 = blockIdx.x * 64;
  const int m0  = blockIdx.y * 128;
  const int zsel = blockIdx.z;
  const float* W = (zsel == 0) ? Wq : (zsel == 1) ? Wk : Wv;

  auto stage = [&](int kk, int buf) {
#pragma unroll
    for (int r = 0; r < 32; ++r) {              // A: 128x32, contiguous in k
      const int e = tid + 128 * r, cl = e & 31, ml = e >> 5;
      lA[buf][ml * LDK + cl] = (_Float16)emb[(size_t)(m0 + ml) * C_DIM + kk + cl];
    }
#pragma unroll
    for (int r = 0; r < 16; ++r) {              // B^T: W[kk+kl][ch0+nl] -> [nl][kl]
      const int e = tid + 128 * r, nl = e & 63, kl = e >> 6;
      lBt[buf][nl * LDK + kl] = (_Float16)W[(size_t)(kk + kl) * CH_DIM + ch0 + nl];
    }
  };

  v8f acc[2][4] = {};
  stage(0, 0);
#pragma unroll
  for (int it = 0; it < 2; ++it) {
    const int buf = it & 1;
    __syncthreads();
    if (it + 1 < 2) stage(32, buf ^ 1);
    v16h a[2], b[4];
#pragma unroll
    for (int s = 0; s < 2; ++s)
      a[s] = frag_lds(&lA[buf][(wave * 32 + s * 16 + (lane & 15)) * LDK], h);
#pragma unroll
    for (int t = 0; t < 4; ++t)
      b[t] = frag_lds(&lBt[buf][(t * 16 + (lane & 15)) * LDK], h);
#pragma unroll
    for (int s = 0; s < 2; ++s)
#pragma unroll
      for (int t = 0; t < 4; ++t)
        acc[s][t] = __builtin_amdgcn_wmma_f32_16x16x32_f16(false, a[s], false, b[t],
                                                           (short)0, acc[s][t], false, false);
  }
  const int nc = lane & 15;
  if (zsel == 2) {                              // V: row-major [m][ch]
    const int mbase = m0 + wave * 32 + h * 8;   // D: vgpr r -> m = r + 8*h
#pragma unroll
    for (int s = 0; s < 2; ++s)
#pragma unroll
      for (int t = 0; t < 4; ++t)
#pragma unroll
        for (int r = 0; r < 8; ++r)
          V16[(size_t)(mbase + s * 16 + r) * CH_DIM + ch0 + t * 16 + nc] = (_Float16)acc[s][t][r];
  } else {                                      // Q/K: transposed [sample][ch][n]
    _Float16* dstT = ((zsel == 0) ? Q16 : K16) + (size_t)(m0 >> 11) * CH_DIM * N_SEQ;
    const int nbase = (m0 & (N_SEQ - 1)) + wave * 32 + h * 8;
#pragma unroll
    for (int s = 0; s < 2; ++s)
#pragma unroll
      for (int t = 0; t < 4; ++t) {
        v8h pk;
#pragma unroll
        for (int r = 0; r < 8; ++r) pk[r] = (_Float16)acc[s][t][r];
        *(v8h*)(dstT + (size_t)(ch0 + t * 16 + nc) * N_SEQ + nbase + s * 16) = pk;
      }
  }
}

// ---------------------------------------------------------------------------
// K2: scores.  S[b,i,j] = sum_n QT[qb+b][i][n] * KT[kb + j/512][j%512][n]
// M=512 (i), N=4096 (j), K=2048 (n).  Block tile 128x64.  grid = (64, 4, 8)
// Both tiles staged via async global->LDS; double-buffered (DMA || WMMA).
// Fused sum / sumsq -> f32 atomics (InstanceNorm stats).
// ---------------------------------------------------------------------------
__global__ __launch_bounds__(128)
void scores_gemm_kernel(const _Float16* __restrict__ Q16, const _Float16* __restrict__ K16,
                        float* __restrict__ S, float* __restrict__ stats,
                        int qbase, int kbase)
{
  __shared__ __align__(16) _Float16 lA[2][128 * LDK];   // [i][n]
  __shared__ __align__(16) _Float16 lBt[2][64 * LDK];   // [j][n]
  __shared__ float red[128];
  const int tid = threadIdx.x, lane = tid & 31, wave = tid >> 5, h = lane >> 4;
  const int j0 = blockIdx.x * 64;
  const int i0 = blockIdx.y * 128;
  const int b  = blockIdx.z;
  const _Float16* QT = Q16 + (size_t)(qbase + b) * CH_DIM * N_SEQ;            // [i][n]
  const _Float16* KT = K16 + (size_t)(kbase + (j0 >> 9)) * CH_DIM * N_SEQ;    // [ch][n]
  const int ch0 = j0 & (CH_DIM - 1);

  auto stage = [&](int n0, int buf) {
    // A tile 128x32: 512 16B chunks, 4 per thread, all async
#pragma unroll
    for (int r = 0; r < 4; ++r) {
      const int c = tid + 128 * r, row = c >> 2, q = (c & 3) << 3;
      async_g2l_b128(&lA[buf][row * LDK + q], QT + (size_t)(i0 + row) * N_SEQ + n0 + q);
    }
    // B tile 64x32: 256 16B chunks, 2 per thread, all async
#pragma unroll
    for (int r = 0; r < 2; ++r) {
      const int c = tid + 128 * r, row = c >> 2, q = (c & 3) << 3;
      async_g2l_b128(&lBt[buf][row * LDK + q], KT + (size_t)(ch0 + row) * N_SEQ + n0 + q);
    }
  };

  v8f acc[2][4] = {};
  stage(0, 0);
  for (int it = 0; it < N_SEQ / 32; ++it) {
    const int buf = it & 1;
    async_wait_all();                // tile it landed in LDS (this wave's copies)
    __syncthreads();                 // ... and everyone else's
    if (it + 1 < N_SEQ / 32) stage((it + 1) * 32, buf ^ 1);   // DMA next || compute cur
    v16h a[2], bf[4];
#pragma unroll
    for (int s = 0; s < 2; ++s)
      a[s] = frag_lds(&lA[buf][(wave * 32 + s * 16 + (lane & 15)) * LDK], h);
#pragma unroll
    for (int t = 0; t < 4; ++t)
      bf[t] = frag_lds(&lBt[buf][(t * 16 + (lane & 15)) * LDK], h);
#pragma unroll
    for (int s = 0; s < 2; ++s)
#pragma unroll
      for (int t = 0; t < 4; ++t)
        acc[s][t] = __builtin_amdgcn_wmma_f32_16x16x32_f16(false, a[s], false, bf[t],
                                                           (short)0, acc[s][t], false, false);
  }
  float s1 = 0.f, s2 = 0.f;
  const int ibase = i0 + wave * 32 + h * 8;
  const int jc    = j0 + (lane & 15);
  float* Sb = S + (size_t)b * 512 * NJ;
#pragma unroll
  for (int s = 0; s < 2; ++s)
#pragma unroll
    for (int t = 0; t < 4; ++t)
#pragma unroll
      for (int r = 0; r < 8; ++r) {
        const float x = acc[s][t][r];
        Sb[(size_t)(ibase + s * 16 + r) * NJ + jc + t * 16] = x;
        s1 += x; s2 += x * x;
      }
  __syncthreads();
  red[tid] = s1; __syncthreads();
  for (int o = 64; o > 0; o >>= 1) { if (tid < o) red[tid] += red[tid + o]; __syncthreads(); }
  const float bs1 = red[0]; __syncthreads();
  red[tid] = s2; __syncthreads();
  for (int o = 64; o > 0; o >>= 1) { if (tid < o) red[tid] += red[tid + o]; __syncthreads(); }
  if (tid == 0) {
    atomicAdd(&stats[b * 2 + 0], bs1);
    atomicAdd(&stats[b * 2 + 1], red[0]);
  }
}

// ---------------------------------------------------------------------------
// K3: InstanceNorm + row softmax, in place in S.  grid = (512, 8), 256 thr.
// ---------------------------------------------------------------------------
__global__ __launch_bounds__(256)
void softmax_kernel(float* __restrict__ S, const float* __restrict__ stats)
{
  __shared__ float red[256];
  const int i = blockIdx.x, b = blockIdx.y, tid = threadIdx.x;
  const float inv = 1.0f / (512.0f * 4096.0f);
  const float mu  = stats[b * 2] * inv;
  const float var = stats[b * 2 + 1] * inv - mu * mu;   // biased var (torch)
  const float rsg = rsqrtf(var + EPS_IN);
  float* row = S + ((size_t)b * 512 + i) * NJ;

  float x[16];
  float mx = -3.4e38f;
#pragma unroll
  for (int t = 0; t < 16; ++t) {
    x[t] = (row[tid + 256 * t] - mu) * rsg;
    mx = fmaxf(mx, x[t]);
  }
  red[tid] = mx; __syncthreads();
  for (int o = 128; o > 0; o >>= 1) { if (tid < o) red[tid] = fmaxf(red[tid], red[tid + o]); __syncthreads(); }
  mx = red[0]; __syncthreads();
  float s = 0.f;
#pragma unroll
  for (int t = 0; t < 16; ++t) { x[t] = __expf(x[t] - mx); s += x[t]; }
  red[tid] = s; __syncthreads();
  for (int o = 128; o > 0; o >>= 1) { if (tid < o) red[tid] += red[tid + o]; __syncthreads(); }
  const float rs = 1.0f / red[0];
#pragma unroll
  for (int t = 0; t < 16; ++t) row[tid + 256 * t] = x[t] * rs;
}

// ---------------------------------------------------------------------------
// K4: ctx.  CTX[b,n,i] = sum_j V16[vb + j/512][n][j%512] * attn[b,i,j]
// M=2048 (n), N=512 (i), K=4096 (j).  Block tile 128x64.  grid = (8, 16, 8)
// A (V16, f16, k-contiguous) async; B (attn f32->f16) cvt; double-buffered.
// ---------------------------------------------------------------------------
__global__ __launch_bounds__(128)
void ctx_gemm_kernel(const float* __restrict__ S, const _Float16* __restrict__ V16,
                     _Float16* __restrict__ CTX, int vbase)
{
  __shared__ __align__(16) _Float16 lA[2][128 * LDK];   // vcat: [n][j]
  __shared__ __align__(16) _Float16 lBt[2][64 * LDK];   // attn: [i][j]
  const int tid = threadIdx.x, lane = tid & 31, wave = tid >> 5, h = lane >> 4;
  const int i0 = blockIdx.x * 64;     // N (CH)
  const int n0 = blockIdx.y * 128;    // M (seq)
  const int b  = blockIdx.z;
  const float* Sb = S + (size_t)b * 512 * NJ;
  _Float16* Cb = CTX + (size_t)b * N_SEQ * CH_DIM;

  auto stage = [&](int j0s, int buf) {
    const _Float16* Vb = V16 + (size_t)(vbase + (j0s >> 9)) * N_SEQ * CH_DIM;
    const int ch0 = j0s & (CH_DIM - 1);
#pragma unroll
    for (int r = 0; r < 4; ++r) {               // A tile 128x32: async
      const int c = tid + 128 * r, row = c >> 2, q = (c & 3) << 3;
      async_g2l_b128(&lA[buf][row * LDK + q], Vb + (size_t)(n0 + row) * CH_DIM + ch0 + q);
    }
#pragma unroll
    for (int r = 0; r < 16; ++r) {              // B^T tile 64x32: f32 attn -> f16
      const int e = tid + 128 * r, jl = e & 31, il = e >> 5;
      lBt[buf][il * LDK + jl] = (_Float16)Sb[(size_t)(i0 + il) * NJ + j0s + jl];
    }
  };

  v8f acc[2][4] = {};
  stage(0, 0);
  for (int it = 0; it < NJ / 32; ++it) {
    const int buf = it & 1;
    async_wait_all();
    __syncthreads();
    if (it + 1 < NJ / 32) stage((it + 1) * 32, buf ^ 1);
    v16h a[2], bf[4];
#pragma unroll
    for (int s = 0; s < 2; ++s)
      a[s] = frag_lds(&lA[buf][(wave * 32 + s * 16 + (lane & 15)) * LDK], h);
#pragma unroll
    for (int t = 0; t < 4; ++t)
      bf[t] = frag_lds(&lBt[buf][(t * 16 + (lane & 15)) * LDK], h);
#pragma unroll
    for (int s = 0; s < 2; ++s)
#pragma unroll
      for (int t = 0; t < 4; ++t)
        acc[s][t] = __builtin_amdgcn_wmma_f32_16x16x32_f16(false, a[s], false, bf[t],
                                                           (short)0, acc[s][t], false, false);
  }
  const int nbase = n0 + wave * 32 + h * 8;
  const int ic    = i0 + (lane & 15);
#pragma unroll
  for (int s = 0; s < 2; ++s)
#pragma unroll
    for (int t = 0; t < 4; ++t)
#pragma unroll
      for (int r = 0; r < 8; ++r)
        Cb[(size_t)(nbase + s * 16 + r) * CH_DIM + ic + t * 16] = (_Float16)acc[s][t][r];
}

// ---------------------------------------------------------------------------
// K5: output projection. out[m, c] = sum_i CTX[m, i] * Wo[i, c]
// M = 8*2048 flat per direction, K = 512, N = 64.  grid = (128)
// A (CTX f16) async; B (Wo f32->f16) cvt; double-buffered.
// ---------------------------------------------------------------------------
__global__ __launch_bounds__(128)
void out_gemm_kernel(const _Float16* __restrict__ CTX, const float* __restrict__ Wo,
                     float* __restrict__ out, int outbase)
{
  __shared__ __align__(16) _Float16 lA[2][128 * LDK];   // [m][i]
  __shared__ __align__(16) _Float16 lBt[2][64 * LDK];   // Wo^T: [c][i]
  const int tid = threadIdx.x, lane = tid & 31, wave = tid >> 5, h = lane >> 4;
  const int m0 = blockIdx.x * 128;

  auto stage = [&](int kk, int buf) {
#pragma unroll
    for (int r = 0; r < 4; ++r) {               // A: async
      const int c = tid + 128 * r, row = c >> 2, q = (c & 3) << 3;
      async_g2l_b128(&lA[buf][row * LDK + q], CTX + (size_t)(m0 + row) * CH_DIM + kk + q);
    }
#pragma unroll
    for (int r = 0; r < 16; ++r) {              // B^T: Wo[kk+il][cl] -> [cl][il]
      const int e = tid + 128 * r, cl = e & 63, il = e >> 6;
      lBt[buf][cl * LDK + il] = (_Float16)Wo[(size_t)(kk + il) * C_DIM + cl];
    }
  };

  v8f acc[2][4] = {};
  stage(0, 0);
  for (int it = 0; it < CH_DIM / 32; ++it) {
    const int buf = it & 1;
    async_wait_all();
    __syncthreads();
    if (it + 1 < CH_DIM / 32) stage((it + 1) * 32, buf ^ 1);
    v16h a[2], bf[4];
#pragma unroll
    for (int s = 0; s < 2; ++s)
      a[s] = frag_lds(&lA[buf][(wave * 32 + s * 16 + (lane & 15)) * LDK], h);
#pragma unroll
    for (int t = 0; t < 4; ++t)
      bf[t] = frag_lds(&lBt[buf][(t * 16 + (lane & 15)) * LDK], h);
#pragma unroll
    for (int s = 0; s < 2; ++s)
#pragma unroll
      for (int t = 0; t < 4; ++t)
        acc[s][t] = __builtin_amdgcn_wmma_f32_16x16x32_f16(false, a[s], false, bf[t],
                                                           (short)0, acc[s][t], false, false);
  }
  const int mbase = m0 + wave * 32 + h * 8;
  const int cc    = lane & 15;
  float* O = out + (size_t)outbase * N_SEQ * C_DIM;
#pragma unroll
  for (int s = 0; s < 2; ++s)
#pragma unroll
    for (int t = 0; t < 4; ++t)
#pragma unroll
      for (int r = 0; r < 8; ++r)
        O[(size_t)(mbase + s * 16 + r) * C_DIM + t * 16 + cc] = acc[s][t][r];
}

__global__ void zero_stats_kernel(float* stats) { stats[threadIdx.x] = 0.f; }

// ---------------------------------------------------------------------------
extern "C" void kernel_launch(void* const* d_in, const int* in_sizes, int n_in,
                              void* d_out, int out_size, void* d_ws, size_t ws_size,
                              hipStream_t stream)
{
  const float* emb = (const float*)d_in[0];
  const float* Wq  = (const float*)d_in[1];
  const float* Wk  = (const float*)d_in[2];
  const float* Wv  = (const float*)d_in[3];
  const float* Wo  = (const float*)d_in[4];
  float* out = (float*)d_out;

  // Workspace layout (~185 MB)
  char* ws = (char*)d_ws;
  size_t off = 0;
  _Float16* Q16 = (_Float16*)(ws + off); off += (size_t)16 * N_SEQ * CH_DIM * 2;  // [s][ch][n]
  _Float16* K16 = (_Float16*)(ws + off); off += (size_t)16 * N_SEQ * CH_DIM * 2;  // [s][ch][n]
  _Float16* V16 = (_Float16*)(ws + off); off += (size_t)16 * N_SEQ * CH_DIM * 2;  // [s][n][ch]
  float*    S   = (float*)   (ws + off); off += (size_t)8 * 512 * NJ * 4;
  _Float16* CTX = (_Float16*)(ws + off); off += (size_t)8 * N_SEQ * CH_DIM * 2;
  float*    stats = (float*) (ws + off); off += 64;
  (void)ws_size; (void)in_sizes; (void)n_in; (void)out_size;

  // Q/K/V projections for all 16 samples (both halves used by both directions)
  proj_gemm_kernel<<<dim3(8, 256, 3), 128, 0, stream>>>(emb, Wq, Wk, Wv, Q16, K16, V16);

  for (int dir = 0; dir < 2; ++dir) {
    // dir 0: out_l2u = attend(q = emb_l[0..7], kv = emb_u[8..15]) -> out 0..7
    // dir 1: out_u2l = attend(q = emb_u[8..15], kv = emb_l[0..7]) -> out 8..15
    const int qbase   = dir ? 8 : 0;
    const int kvbase  = dir ? 0 : 8;
    const int outbase = dir ? 8 : 0;

    zero_stats_kernel<<<1, 16, 0, stream>>>(stats);
    scores_gemm_kernel<<<dim3(64, 4, 8), 128, 0, stream>>>(Q16, K16, S, stats, qbase, kvbase);
    softmax_kernel<<<dim3(512, 8), 256, 0, stream>>>(S, stats);
    ctx_gemm_kernel<<<dim3(8, 16, 8), 128, 0, stream>>>(S, V16, CTX, kvbase);
    out_gemm_kernel<<<128, 128, 0, stream>>>(CTX, Wo, out, outbase);
  }
}
